// RevProcessor_44384192037441
// MI455X (gfx1250) — matrix operand order repaired and checked by reading, not compiled, not run on hardware
//
#include <hip/hip_runtime.h>
#include <hip/hip_bf16.h>

typedef __attribute__((ext_vector_type(16))) _Float16 v16h;
typedef __attribute__((ext_vector_type(8)))  float    v8f;

#define LATENT 128
#define GC 64
#define EPSLN 1e-5f
#define NEG_SLOPE 0.2f

// ---- monotonic float<->uint key for atomic max over floats ----
__device__ __forceinline__ unsigned fkey(float f) {
    unsigned u = __float_as_uint(f);
    return (u & 0x80000000u) ? ~u : (u | 0x80000000u);
}
__device__ __forceinline__ float fdec(unsigned k) {
    return __uint_as_float((k & 0x80000000u) ? (k & 0x7FFFFFFFu) : ~k);
}

__global__ void k_zero(float* p, long n) {
    long i = (long)blockIdx.x * blockDim.x + threadIdx.x;
    long st = (long)gridDim.x * blockDim.x;
    for (; i < n; i += st) p[i] = 0.0f;
}

// loop_attr accumulation: per (edge, 4-channel chunk): scatter-add edge_attr into dst rows; count degree
__global__ void k_loopattr(const int* __restrict__ dstA, const float* __restrict__ ea,
                           float* __restrict__ la, float* __restrict__ deg, int E) {
    long idx = (long)blockIdx.x * blockDim.x + threadIdx.x;
    if (idx >= (long)E * 16) return;
    int e = (int)(idx >> 4), q = (int)(idx & 15);
    int d = dstA[e];
    const float4 v = *(const float4*)(ea + (size_t)e * GC + q * 4);
    float* o = la + (size_t)d * GC + q * 4;
    atomicAdd(o + 0, v.x); atomicAdd(o + 1, v.y);
    atomicAdd(o + 2, v.z); atomicAdd(o + 3, v.w);
    if (q == 0) atomicAdd(deg + d, 1.0f);
}

__global__ void k_loopdiv(float* __restrict__ la, const float* __restrict__ deg, int Nn) {
    long idx = (long)blockIdx.x * blockDim.x + threadIdx.x;
    if (idx >= (long)Nn * GC) return;
    int n = (int)(idx >> 6);
    la[idx] /= fmaxf(deg[n], 1.0f);
}

// veA[h][k] = sum_c att_edge[h,c] * We[h*GC+c, k]   (collapses the E x 128 edge GEMM)
__global__ void k_veA(const float* __restrict__ ae, const float* __restrict__ We, float* __restrict__ veA) {
    int t = threadIdx.x;            // 128 threads
    int hh = t >> 6, k = t & 63;
    float s = 0.f;
    for (int c = 0; c < GC; ++c) s += ae[hh * GC + c] * We[(size_t)(hh * GC + c) * GC + k];
    veA[t] = s;
}

// h = y_in @ W^T  via v_wmma_f32_16x16x32_f16. One wave per 16x16 output tile.
// yin: base already offset to the active half; row stride LATENT. Wc: [128,64].
__global__ void k_gemm_wmma(const float* __restrict__ yin, const float* __restrict__ Wc,
                            float* __restrict__ h) {
    int lane  = threadIdx.x & 31;
    int otile = threadIdx.x >> 5;          // 8 waves -> 8 output-col tiles (128/16)
    int hi    = lane >> 4;                 // lane half
    int mrow  = blockIdx.x * 16 + (lane & 15);
    int ocol  = otile * 16 + (lane & 15);
    const float* arow = yin + (size_t)mrow * LATENT;
    v8f c = {};
    #pragma unroll
    for (int kb = 0; kb < GC; kb += 32) {
        v16h a, b;
        #pragma unroll
        for (int e = 0; e < 8; ++e) {      // A 16x32 f16: contiguous k = 16*hi + e, same order as B
            a[e]     = (_Float16)arow[kb + hi * 16 + e];
            a[8 + e] = (_Float16)arow[kb + hi * 16 + 8 + e];
        }
        const float* wrow = Wc + (size_t)ocol * GC + kb + hi * 16;  // B 32x16: lanes0-15 K=0..15
        #pragma unroll
        for (int e = 0; e < 16; ++e) b[e] = (_Float16)wrow[e];
        c = __builtin_amdgcn_wmma_f32_16x16x32_f16(false, a, false, b, (short)0, c, false, false);
    }
    #pragma unroll
    for (int r = 0; r < 8; ++r) {          // C/D tile layout: M = r + 8*hi, N = lane&15
        int m = blockIdx.x * 16 + r + hi * 8;
        h[(size_t)m * LATENT + ocol] = c[r];
    }
}

// alpha_src / alpha_dst per (node, head)
__global__ void k_alphas(const float* __restrict__ h, const float* __restrict__ as,
                         const float* __restrict__ ad, float* __restrict__ aso,
                         float* __restrict__ ado, int Nn) {
    long idx = (long)blockIdx.x * blockDim.x + threadIdx.x;
    if (idx >= (long)Nn * 2) return;
    int n = (int)(idx >> 1), hh = (int)(idx & 1);
    const float* hr = h + (size_t)n * LATENT + hh * GC;
    float s = 0.f, d = 0.f;
    for (int c = 0; c < GC; ++c) { s += hr[c] * as[hh * GC + c]; d += hr[c] * ad[hh * GC + c]; }
    aso[idx] = s; ado[idx] = d;
}

// per edge (incl. self loops): alpha = lrelu(as[src]+ad[dst]+ea.veA); atomic-max per dst
__global__ void k_alpha_edges(const int* __restrict__ srcA, const int* __restrict__ dstA,
                              const float* __restrict__ ea, const float* __restrict__ la,
                              const float* __restrict__ aso, const float* __restrict__ ado,
                              const float* __restrict__ veA, float* __restrict__ ab,
                              unsigned* __restrict__ amx, int E, int ET) {
    long e = (long)blockIdx.x * blockDim.x + threadIdx.x;
    if (e >= ET) return;
    int s, d; const float* eap;
    if (e < E) { s = srcA[e]; d = dstA[e]; eap = ea + (size_t)e * GC; }
    else       { s = (int)(e - E); d = s;  eap = la + (size_t)s * GC; }
    float d0 = 0.f, d1 = 0.f;
    for (int k = 0; k < GC; k += 4) {
        float4 v  = *(const float4*)(eap + k);
        float4 u0 = *(const float4*)(veA + k);
        float4 u1 = *(const float4*)(veA + GC + k);
        d0 += v.x * u0.x + v.y * u0.y + v.z * u0.z + v.w * u0.w;
        d1 += v.x * u1.x + v.y * u1.y + v.z * u1.z + v.w * u1.w;
    }
    float a0 = aso[s * 2 + 0] + ado[d * 2 + 0] + d0;
    float a1 = aso[s * 2 + 1] + ado[d * 2 + 1] + d1;
    a0 = a0 > 0.f ? a0 : NEG_SLOPE * a0;
    a1 = a1 > 0.f ? a1 : NEG_SLOPE * a1;
    ab[e * 2 + 0] = a0; ab[e * 2 + 1] = a1;
    atomicMax(amx + d * 2 + 0, fkey(a0));
    atomicMax(amx + d * 2 + 1, fkey(a1));
}

__global__ void k_exp(const int* __restrict__ dstA, const unsigned* __restrict__ amx,
                      float* __restrict__ ab, float* __restrict__ den, int E, int ET) {
    long idx = (long)blockIdx.x * blockDim.x + threadIdx.x;
    if (idx >= (long)ET * 2) return;
    int e = (int)(idx >> 1), hh = (int)(idx & 1);
    int d = (e < E) ? dstA[e] : (e - E);
    float ex = __expf(ab[idx] - fdec(amx[d * 2 + hh]));
    ab[idx] = ex;
    atomicAdd(den + d * 2 + hh, ex);
}

// aggregate: fbuf[dst] += 0.5*(w0*h[src,head0] + w1*h[src,head1]) per channel
__global__ void k_aggr(const int* __restrict__ srcA, const int* __restrict__ dstA,
                       const float* __restrict__ ab, const float* __restrict__ den,
                       const float* __restrict__ h, float* __restrict__ fbuf, int E, int ET) {
    long idx = (long)blockIdx.x * blockDim.x + threadIdx.x;
    if (idx >= (long)ET * 16) return;
    int e = (int)(idx >> 4), q = (int)(idx & 15);
    int s, d;
    if (e < E) { s = srcA[e]; d = dstA[e]; } else { s = e - E; d = s; }
    float w0 = 0.5f * ab[e * 2 + 0] / den[d * 2 + 0];
    float w1 = 0.5f * ab[e * 2 + 1] / den[d * 2 + 1];
    const float* hr = h + (size_t)s * LATENT + q * 4;
    float4 h0 = *(const float4*)(hr);
    float4 h1 = *(const float4*)(hr + GC);
    float* o = fbuf + (size_t)d * GC + q * 4;
    atomicAdd(o + 0, w0 * h0.x + w1 * h1.x);
    atomicAdd(o + 1, w0 * h0.y + w1 * h1.y);
    atomicAdd(o + 2, w0 * h0.z + w1 * h1.z);
    atomicAdd(o + 3, w0 * h0.w + w1 * h1.w);
}

// bias + LayerNorm + residual write into half `out_half` of x
__global__ void k_ln_add(const float* __restrict__ fbuf, const float* __restrict__ bias,
                         const float* __restrict__ gam, const float* __restrict__ bet,
                         float* __restrict__ xo, int out_half, int Nn) {
    int n = blockIdx.x * blockDim.x + threadIdx.x;
    if (n >= Nn) return;
    const float* f = fbuf + (size_t)n * GC;
    float s = 0.f, s2 = 0.f;
    for (int c = 0; c < GC; ++c) { float v = f[c] + bias[c]; s += v; s2 += v * v; }
    float mu = s * (1.0f / GC);
    float var = fmaxf(s2 * (1.0f / GC) - mu * mu, 0.f);
    float r = rsqrtf(var + EPSLN);
    float* xr = xo + (size_t)n * LATENT + out_half * GC;
    for (int c = 0; c < GC; ++c) {
        float v = f[c] + bias[c];
        xr[c] = xr[c] + (v - mu) * r * gam[c] + bet[c];
    }
}

extern "C" void kernel_launch(void* const* d_in, const int* in_sizes, int n_in,
                              void* d_out, int out_size, void* d_ws, size_t ws_size,
                              hipStream_t stream) {
    const float* x   = (const float*)d_in[0];
    const int*   ei  = (const int*)d_in[1];
    const float* ea  = (const float*)d_in[2];
    const float* W   = (const float*)d_in[4];
    const float* We  = (const float*)d_in[5];
    const float* as  = (const float*)d_in[6];
    const float* ad  = (const float*)d_in[7];
    const float* ae  = (const float*)d_in[8];
    const float* bs  = (const float*)d_in[9];
    const float* gm  = (const float*)d_in[10];
    const float* bt  = (const float*)d_in[11];
    float* xo = (float*)d_out;

    const int Nn = in_sizes[0] / LATENT;      // 50000
    const int E  = in_sizes[1] / 2;           // 1600000
    const int ET = E + Nn;
    const int* srcA = ei;
    const int* dstA = ei + E;

    // workspace layout (floats)
    float* ws   = (float*)d_ws;
    float* h    = ws;                                  // N*128
    float* la   = h + (size_t)Nn * LATENT;             // N*64
    float* deg  = la + (size_t)Nn * GC;                // N
    float* aso  = deg + Nn;                            // N*2
    float* ado  = aso + (size_t)Nn * 2;                // N*2
    unsigned* amx = (unsigned*)(ado + (size_t)Nn * 2); // N*2 (uint keys)
    float* den  = (float*)amx + (size_t)Nn * 2;        // N*2
    float* fbuf = den + (size_t)Nn * 2;                // N*64
    float* ab   = fbuf + (size_t)Nn * GC;              // ET*2
    float* veA  = ab + (size_t)ET * 2;                 // 128

    const int T = 256;
    auto cdiv = [](long a, long b) { return (int)((a + b - 1) / b); };

    // x -> output buffer (updated in place, halves alternate)
    hipMemcpyAsync(xo, x, (size_t)Nn * LATENT * sizeof(float), hipMemcpyDeviceToDevice, stream);

    // loop_attr (fill_value='mean') depends only on edge_attr/dst: compute once
    k_zero<<<4096, T, 0, stream>>>(la, (long)Nn * GC + Nn);  // la + deg contiguous
    k_loopattr<<<cdiv((long)E * 16, T), T, 0, stream>>>(dstA, ea, la, deg, E);
    k_loopdiv<<<cdiv((long)Nn * GC, T), T, 0, stream>>>(la, deg, Nn);

    for (int conv = 0; conv < 4; ++conv) {
        int g = conv & 1;
        int in_half = (g == 0) ? 1 : 0;
        int out_half = g;
        // reset amx(=0 is minimal key) + den + fbuf (contiguous N*68 floats)
        k_zero<<<4096, T, 0, stream>>>((float*)amx, (long)Nn * 68);
        k_veA<<<1, 128, 0, stream>>>(ae + conv * 128, We + (size_t)conv * LATENT * GC, veA);
        k_gemm_wmma<<<Nn / 16, 256, 0, stream>>>(xo + in_half * GC,
                                                 W + (size_t)conv * LATENT * GC, h);
        k_alphas<<<cdiv((long)Nn * 2, T), T, 0, stream>>>(h, as + conv * 128, ad + conv * 128,
                                                          aso, ado, Nn);
        k_alpha_edges<<<cdiv(ET, T), T, 0, stream>>>(srcA, dstA, ea, la, aso, ado, veA,
                                                     ab, amx, E, ET);
        k_exp<<<cdiv((long)ET * 2, T), T, 0, stream>>>(dstA, amx, ab, den, E, ET);
        k_aggr<<<cdiv((long)ET * 16, T), T, 0, stream>>>(srcA, dstA, ab, den, h, fbuf, E, ET);
        k_ln_add<<<cdiv(Nn, 128), 128, 0, stream>>>(fbuf, bs + conv * GC, gm + conv * GC,
                                                    bt + conv * GC, xo, out_half, Nn);
    }
}
